// EarthAttention3D_65034394796760
// MI455X (gfx1250) — matrix operand order repaired and checked by reading, not compiled, not run on hardware
//
#include <hip/hip_runtime.h>
#include <hip/hip_bf16.h>

// EarthAttention3D for MI455X (gfx1250, wave32, WMMA).
// f16 operands / f32 accumulation via v_wmma_f32_16x16x32_f16:
//  ~60 GFLOP vs ~320MB HBM traffic -> fp32 matrix math would be compute-bound
//  by >10x; f16 WMMA pushes toward the 23.3 TB/s memory roof.
// Staging copies use CDNA5 GLOBAL_LOAD_ASYNC_TO_LDS_B128 (ASYNCcnt) when the
// toolchain exposes the builtins, else fall back to b128 VGPR copies.

typedef __attribute__((ext_vector_type(16))) _Float16 v16h;
typedef __attribute__((ext_vector_type(8)))  _Float16 v8h;
typedef __attribute__((ext_vector_type(8)))  float    v8f;

#define WIN_TOTAL 1024
#define NTOK 144          // tokens per window (9 tiles of 16)
#define DIM 192           // 6 K-chunks of 32
#define HEADS 6
#define DH 32
#define NPAD 160          // token dim padded to 5 K-chunks of 32 for AV
#define QSCALE 0.17677669529663687f   // 32^-0.5

#if defined(__has_builtin)
#if __has_builtin(__builtin_amdgcn_global_load_async_to_lds_b128) && \
    __has_builtin(__builtin_amdgcn_s_wait_asynccnt)
#define USE_ASYNC_LDS 1
#endif
#endif

// ---- staging helpers -------------------------------------------------------

// Builtin signature (from hipcc diagnostic): first arg is int4* in the global
// address space, second is int4* in LDS, then imm offset / imm cpol.
typedef int v4i_vs __attribute__((vector_size(16)));

__device__ __forceinline__ void copy_g2l_b128(void* lds_dst, const void* gsrc) {
#ifdef USE_ASYNC_LDS
  __builtin_amdgcn_global_load_async_to_lds_b128(
      (__attribute__((address_space(1))) v4i_vs*)gsrc,
      (__attribute__((address_space(3))) v4i_vs*)lds_dst,
      /*offset=*/0, /*cpol=*/0);
#else
  *(uint4*)lds_dst = *(const uint4*)gsrc;
#endif
}

__device__ __forceinline__ void async_fence() {
#ifdef USE_ASYNC_LDS
  __builtin_amdgcn_s_wait_asynccnt(0);
#endif
}

// ---- WMMA helpers ----------------------------------------------------------

__device__ __forceinline__ v8f wmma16(v16h a, v16h b, v8f c) {
  // D(16x16,f32) = A(16x32,f16) * B(32x16,f16) + C
  return __builtin_amdgcn_wmma_f32_16x16x32_f16(
      /*neg_a=*/false, a, /*neg_b=*/false, b,
      /*c_mod=*/(short)0, c, /*reuse_a=*/false, /*reuse_b=*/false);
}

// A-matrix 16x32 f16 fragment. Lane group (hi = lane>>4) holds
// K = {hi*8+0..7, hi*8+16..23} of row (m0 + (lane&15)); rowbase is the start
// of that row (K-contiguous storage), k0 = 32-aligned chunk offset.
__device__ __forceinline__ v16h load_a(const _Float16* rowbase, int k0, int hi) {
  v8h lo = *(const v8h*)(rowbase + k0 + hi * 8);
  v8h hh = *(const v8h*)(rowbase + k0 + hi * 8 + 16);
  v16h r;
#pragma unroll
  for (int i = 0; i < 8; ++i) { r[i] = lo[i]; r[i + 8] = hh[i]; }
  return r;
}

// B-matrix 32x16 f16 fragment. Lane (lane&15) owns output column n; lane
// group hi holds K = {hi*8+0..7, hi*8+16..23}. colbase is the K-contiguous storage of
// that column (transposed weights / K rows / Vt rows).
__device__ __forceinline__ v16h load_b(const _Float16* colbase, int k0, int hi) {
  v8h lo = *(const v8h*)(colbase + k0 + hi * 8);
  v8h hh = *(const v8h*)(colbase + k0 + hi * 8 + 16);
  v16h r;
#pragma unroll
  for (int i = 0; i < 8; ++i) { r[i] = lo[i]; r[i + 8] = hh[i]; }
  return r;
}

// ---- Kernel 1a: weights -> f16, transposed (column-major, K-contiguous) ----

__global__ void __launch_bounds__(256)
cvt_weights(const float* __restrict__ w1, const float* __restrict__ w2,
            _Float16* __restrict__ w1t, _Float16* __restrict__ w2t) {
  int g = blockIdx.x * 256 + threadIdx.x;
  if (g < 576 * 192) {                      // w1t[c][k] = w1[k][c]
    int c = g / 192, k = g - c * 192;
    w1t[g] = (_Float16)w1[k * 576 + c];
  } else if (g < 576 * 192 + 192 * 192) {   // w2t[c][k] = w2[k][c]
    int g2 = g - 576 * 192;
    int c = g2 / 192, k = g2 - c * 192;
    w2t[g2] = (_Float16)w2[k * 192 + c];
  }
}

// ---- Kernel 1b: x -> f16 once (avoids 9x redundant conversion in QKV) ------

__global__ void __launch_bounds__(256)
cvt_x(const float* __restrict__ x, _Float16* __restrict__ xh, int n8) {
  int g = blockIdx.x * 256 + threadIdx.x;
  if (g >= n8) return;
  const float4* xg = (const float4*)x;
  float4 f0 = xg[g * 2], f1 = xg[g * 2 + 1];
  _Float16 h[8] = {(_Float16)f0.x, (_Float16)f0.y, (_Float16)f0.z, (_Float16)f0.w,
                   (_Float16)f1.x, (_Float16)f1.y, (_Float16)f1.z, (_Float16)f1.w};
  *(uint4*)(xh + (size_t)g * 8) = *(const uint4*)h;
}

// ---- Kernel 2: QKV projection ---------------------------------------------
// grid (1024 windows, 9 slabs of 64 output cols), 128 threads = 4 waves.
// Each wave owns one 16-col strip: 9 M-tiles x 6 K-chunks = 54 WMMA.
// Epilogue: +b1, scale Q; q/k stored row-major [tok][dh], v transposed
// [dh][tok] so attention B-loads are K-contiguous.

__global__ void __launch_bounds__(128)
qkv_gemm(const _Float16* __restrict__ xh, const _Float16* __restrict__ w1t,
         const float* __restrict__ b1, _Float16* __restrict__ qws,
         _Float16* __restrict__ kws, _Float16* __restrict__ vtws) {
  extern __shared__ char smem[];
  _Float16* Xh = (_Float16*)smem;                        // 144*192 f16
  _Float16* Wt = (_Float16*)(smem + NTOK * DIM * 2);     // 64*192 f16

  const int win = blockIdx.x, slab = blockIdx.y, tid = threadIdx.x;

  // stage X (already f16) via async b128 copies
  const uint4* xg = (const uint4*)(xh + (size_t)win * (NTOK * DIM));
  for (int i = tid; i < NTOK * DIM / 8; i += 128)
    copy_g2l_b128((uint4*)Xh + i, xg + i);
  // stage 64-column slab of transposed w1 (K-contiguous rows)
  const uint4* wg = (const uint4*)(w1t + (size_t)slab * 64 * DIM);
  for (int i = tid; i < 64 * DIM / 8; i += 128)
    copy_g2l_b128((uint4*)Wt + i, wg + i);
  async_fence();
  __syncthreads();

  const int wave = tid >> 5, lane = tid & 31, l = lane & 15, hi = lane >> 4;
  const int n0 = wave * 16;

  v8f acc[9];
#pragma unroll
  for (int t = 0; t < 9; ++t)
#pragma unroll
    for (int i = 0; i < 8; ++i) acc[t][i] = 0.f;

  for (int kc = 0; kc < 6; ++kc) {
    int k0 = kc * 32;
    v16h b = load_b(Wt + (n0 + l) * DIM, k0, hi);
#pragma unroll
    for (int mt = 0; mt < 9; ++mt) {
      v16h a = load_a(Xh + (mt * 16 + l) * DIM, k0, hi);
      acc[mt] = wmma16(a, b, acc[mt]);
    }
  }

  // epilogue: C/D layout -> (M = v + 8*hi, N = l)
  const int gc = slab * 64 + n0 + l;         // 0..575 : [3][HEADS][DH]
  const int part = gc / 192;
  const int rem = gc - part * 192;
  const int h = rem >> 5, d = rem & 31;
  const float bv = b1[gc];
  const float mul = (part == 0) ? QSCALE : 1.f;
  const size_t hb = ((size_t)win * HEADS + h);
#pragma unroll
  for (int mt = 0; mt < 9; ++mt) {
#pragma unroll
    for (int v = 0; v < 8; ++v) {
      int m = mt * 16 + v + hi * 8;
      float val = (acc[mt][v] + bv) * mul;
      if (part == 0)       qws[(hb * NTOK + m) * DH + d] = (_Float16)val;
      else if (part == 1)  kws[(hb * NTOK + m) * DH + d] = (_Float16)val;
      else                 vtws[(hb * DH + d) * NTOK + m] = (_Float16)val;
    }
  }
}

// ---- Kernel 3: fused attention per (window, head) -------------------------
// S = Q K^T (one 16x16x32 WMMA per tile, DH==32) + earth bias gather + mask,
// f32 row softmax, P(f16, padded to 160 cols) @ V over 5 K-chunks.
// 256 threads = 8 waves; ~163KB dynamic LDS (within 320KB/WG).

__global__ void __launch_bounds__(256)
attention(const _Float16* __restrict__ qws, const _Float16* __restrict__ kws,
          const _Float16* __restrict__ vtws, const float* __restrict__ mask,
          const float* __restrict__ bias_table, const int* __restrict__ pos,
          _Float16* __restrict__ yws) {
  extern __shared__ char smem[];
  _Float16* Qh = (_Float16*)smem;                        //  9216 B
  _Float16* Kh = (_Float16*)(smem + 9216);               //  9216 B
  _Float16* Vt = (_Float16*)(smem + 18432);              // 32*160 f16 = 10240 B
  float*    S  = (float*)   (smem + 28672);              // 144*160 f32 = 92160 B
  _Float16* P  = (_Float16*)(smem + 120832);             // 144*160 f16 = 46080 B

  const int win = blockIdx.x, h = blockIdx.y, tid = threadIdx.x;
  const size_t base = ((size_t)win * HEADS + h) * (NTOK * DH);

  // async-stage Q and K (b128 chunks)
  const uint4* qg = (const uint4*)(qws + base);
  const uint4* kg = (const uint4*)(kws + base);
  for (int i = tid; i < NTOK * DH / 8; i += 256) {
    copy_g2l_b128((uint4*)Qh + i, qg + i);
    copy_g2l_b128((uint4*)Kh + i, kg + i);
  }
  // V transposed [dh][tok], zero-padded to 160 tokens
  const _Float16* vg = vtws + base;
  for (int i = tid; i < DH * NPAD; i += 256) {
    int d = i / NPAD, m = i - d * NPAD;
    Vt[i] = (m < NTOK) ? vg[d * NTOK + m] : (_Float16)0.f;
  }
  async_fence();
  __syncthreads();

  const int wave = tid >> 5, lane = tid & 31, l = lane & 15, hi = lane >> 4;
  const int wgrp = win >> 4;               // lat/lev window group (64 of them)

  // --- scores: 81 tiles of 16x16 over 8 waves
  for (int tt = wave; tt < 81; tt += 8) {
    int nt = tt / 9, mt = tt - nt * 9;
    v8f acc;
#pragma unroll
    for (int i = 0; i < 8; ++i) acc[i] = 0.f;
    v16h a = load_a(Qh + (nt * 16 + l) * DH, 0, hi);
    v16h b = load_b(Kh + (mt * 16 + l) * DH, 0, hi);
    acc = wmma16(a, b, acc);
    int m = mt * 16 + l;
#pragma unroll
    for (int v = 0; v < 8; ++v) {
      int n = nt * 16 + v + hi * 8;
      int pidx = pos[n * NTOK + m];
      float bv = bias_table[((size_t)pidx * 64 + wgrp) * HEADS + h];
      float mv = mask[((size_t)win * NTOK + n) * NTOK + m];
      S[n * NPAD + m] = acc[v] + bv + mv;
    }
  }
  __syncthreads();

  // --- f32 row softmax (one thread per row), pack P to f16 with zero pad
  if (tid < NTOK) {
    float* Sr = S + tid * NPAD;
    float mx = -3.4e38f;
    for (int m = 0; m < NTOK; ++m) mx = fmaxf(mx, Sr[m]);
    float sum = 0.f;
    for (int m = 0; m < NTOK; ++m) { float e = __expf(Sr[m] - mx); Sr[m] = e; sum += e; }
    float inv = 1.f / sum;
    _Float16* Pr = P + tid * NPAD;
    for (int m = 0; m < NTOK; ++m) Pr[m] = (_Float16)(Sr[m] * inv);
    for (int m = NTOK; m < NPAD; ++m) Pr[m] = (_Float16)0.f;
  }
  __syncthreads();

  // --- O = P @ V : 9 n-tiles x 2 dh-tiles, K = 160 (5 chunks)
  for (int tt = wave; tt < 18; tt += 8) {
    int nt = tt >> 1, dt = tt & 1;
    v8f acc;
#pragma unroll
    for (int i = 0; i < 8; ++i) acc[i] = 0.f;
#pragma unroll
    for (int kc = 0; kc < 5; ++kc) {
      v16h a = load_a(P + (nt * 16 + l) * NPAD, kc * 32, hi);
      v16h b = load_b(Vt + (dt * 16 + l) * NPAD, kc * 32, hi);
      acc = wmma16(a, b, acc);
    }
    int dcol = dt * 16 + l;
#pragma unroll
    for (int v = 0; v < 8; ++v) {
      int n = nt * 16 + v + hi * 8;
      yws[((size_t)win * NTOK + n) * DIM + h * DH + dcol] = (_Float16)acc[v];
    }
  }
}

// ---- Kernel 4: output projection + b2, f32 store --------------------------
// grid 1024, 128 threads = 4 waves, each wave owns 3 column tiles.

__global__ void __launch_bounds__(128)
proj(const _Float16* __restrict__ yws, const _Float16* __restrict__ w2t,
     const float* __restrict__ b2, float* __restrict__ out) {
  extern __shared__ char smem[];
  _Float16* Yh = (_Float16*)smem;                        // 144*192 f16
  _Float16* Wt = (_Float16*)(smem + NTOK * DIM * 2);     // 192*192 f16

  const int win = blockIdx.x, tid = threadIdx.x;
  const uint4* yg = (const uint4*)(yws + (size_t)win * NTOK * DIM);
  for (int i = tid; i < NTOK * DIM / 8; i += 128)
    copy_g2l_b128((uint4*)Yh + i, yg + i);
  const uint4* wg = (const uint4*)w2t;
  for (int i = tid; i < DIM * DIM / 8; i += 128)
    copy_g2l_b128((uint4*)Wt + i, wg + i);
  async_fence();
  __syncthreads();

  const int wave = tid >> 5, lane = tid & 31, l = lane & 15, hi = lane >> 4;
  for (int ct = wave; ct < 12; ct += 4) {
    v8f acc[9];
#pragma unroll
    for (int t = 0; t < 9; ++t)
#pragma unroll
      for (int i = 0; i < 8; ++i) acc[t][i] = 0.f;
    for (int kc = 0; kc < 6; ++kc) {
      int k0 = kc * 32;
      v16h b = load_b(Wt + (ct * 16 + l) * DIM, k0, hi);
#pragma unroll
      for (int mt = 0; mt < 9; ++mt) {
        v16h a = load_a(Yh + (mt * 16 + l) * DIM, k0, hi);
        acc[mt] = wmma16(a, b, acc[mt]);
      }
    }
    int col = ct * 16 + l;
    float bb = b2[col];
#pragma unroll
    for (int mt = 0; mt < 9; ++mt)
#pragma unroll
      for (int v = 0; v < 8; ++v) {
        int n = mt * 16 + v + hi * 8;
        out[((size_t)win * NTOK + n) * DIM + col] = acc[mt][v] + bb;
      }
  }
}

// ---- host launch -----------------------------------------------------------

extern "C" void kernel_launch(void* const* d_in, const int* in_sizes, int n_in,
                              void* d_out, int out_size, void* d_ws, size_t ws_size,
                              hipStream_t stream) {
  (void)in_sizes; (void)n_in; (void)out_size; (void)ws_size;
  const float* x          = (const float*)d_in[0];
  const float* mask       = (const float*)d_in[1];
  const float* w1         = (const float*)d_in[2];
  const float* b1         = (const float*)d_in[3];
  const float* w2         = (const float*)d_in[4];
  const float* b2         = (const float*)d_in[5];
  const float* bias_table = (const float*)d_in[6];
  const int*   pos        = (const int*)d_in[7];

  char* ws = (char*)d_ws;
  size_t off = 0;
  auto take = [&](size_t bytes) -> char* {
    char* p = ws + off;
    off = (off + bytes + 255) & ~(size_t)255;
    return p;
  };
  _Float16* w1t  = (_Float16*)take((size_t)576 * 192 * 2);
  _Float16* w2t  = (_Float16*)take((size_t)192 * 192 * 2);
  _Float16* xh   = (_Float16*)take((size_t)WIN_TOTAL * NTOK * DIM * 2);
  _Float16* qws  = (_Float16*)take((size_t)WIN_TOTAL * HEADS * NTOK * DH * 2);
  _Float16* kws  = (_Float16*)take((size_t)WIN_TOTAL * HEADS * NTOK * DH * 2);
  _Float16* vtws = (_Float16*)take((size_t)WIN_TOTAL * HEADS * DH * NTOK * 2);
  _Float16* yws  = (_Float16*)take((size_t)WIN_TOTAL * NTOK * DIM * 2);

  cvt_weights<<<576, 256, 0, stream>>>(w1, w2, w1t, w2t);

  const int n8 = WIN_TOTAL * NTOK * DIM / 8;       // 3,538,944 uint4 stores
  cvt_x<<<(n8 + 255) / 256, 256, 0, stream>>>(x, xh, n8);

  size_t lds_qkv = (size_t)NTOK * DIM * 2 + (size_t)64 * DIM * 2;     // 79872 B
  qkv_gemm<<<dim3(WIN_TOTAL, 9), 128, lds_qkv, stream>>>(xh, w1t, b1, qws, kws, vtws);

  size_t lds_att = 120832 + (size_t)NTOK * NPAD * 2;                  // 166912 B
  attention<<<dim3(WIN_TOTAL, HEADS), 256, lds_att, stream>>>(
      qws, kws, vtws, mask, bias_table, pos, yws);

  size_t lds_prj = (size_t)NTOK * DIM * 2 + (size_t)DIM * DIM * 2;    // 129024 B
  proj<<<WIN_TOTAL, 128, lds_prj, stream>>>(yws, w2t, b2, (float*)d_out);
}
